// ConvLSTMEncoder_84808424226887
// MI455X (gfx1250) — matrix operand order repaired and checked by reading, not compiled, not run on hardware
//
#include <hip/hip_runtime.h>
#include <hip/hip_bf16.h>
#include <math.h>
#include <stdint.h>

typedef __attribute__((ext_vector_type(16))) _Float16 v16h;
typedef __attribute__((ext_vector_type(8)))  _Float16 v8h;
typedef __attribute__((ext_vector_type(8)))  float    v8f;

// Global (address_space(1)) pointer types: keep weight loads on the
// global_load path (LOADcnt only), not flat (LOADcnt+DScnt coupling).
typedef const __attribute__((address_space(1))) _Float16* gwhalf;
typedef const __attribute__((address_space(1))) v16h*     gv16hp;

#define B_    256
#define T_    512
#define CIN   64
#define H     128
#define NG    512                 // 4*H gate outputs per layer
#define ACT_STRIDE 456            // 448 cols (x|h1|h2|h3) padded to kill LDS bank conflicts
#define CC_STRIDE  516            // 512 padded
#define FRAG_H 512                // f16 elements per 32x16 B-fragment (1 KB)
#define NK1 6                     // 192/32 K-tiles layer 1
#define NK23 8                    // 256/32 K-tiles layers 2,3
#define WS_L1 0
#define WS_L2 (32*NK1*FRAG_H)             // 98304 halves
#define WS_L3 (WS_L2 + 32*NK23*FRAG_H)    // 229376 halves
#define WS_TOT (WS_L3 + 32*NK23*FRAG_H)   // 360448 halves (704 KB f16)

// ---------------------------------------------------------------------------
// Prep: extract conv center tap W[:,:,1], convert to f16, and lay out as WMMA
// B-fragments (K=32 x N=16). Fragment element (lane l, half hh) holds
// Wc[N = ntile*16 + (l&15)][K = ktile*32 + (l>=16?16:0) + hh], lane-major so
// each lane in the main kernel reads 32 contiguous bytes.
// ---------------------------------------------------------------------------
__global__ void prep_weights_kernel(const float* __restrict__ W1,
                                    const float* __restrict__ W2,
                                    const float* __restrict__ W3,
                                    _Float16* __restrict__ ws)
{
    int i = blockIdx.x * blockDim.x + threadIdx.x;
    if (i >= WS_TOT) return;
    const float* W; int kdim, nk, base;
    if (i < WS_L2)      { W = W1; kdim = CIN + H; nk = NK1;  base = WS_L1; }
    else if (i < WS_L3) { W = W2; kdim = 2 * H;   nk = NK23; base = WS_L2; }
    else                { W = W3; kdim = 2 * H;   nk = NK23; base = WS_L3; }
    int li = i - base;
    int f  = li >> 9;            // fragment index within layer
    int r  = li & 511;
    int l  = r >> 4;             // lane 0..31
    int hh = r & 15;             // half within lane
    int ntile = f / nk, ktile = f % nk;
    int N  = ntile * 16 + (l & 15);
    int Kk = ktile * 32 + ((l >= 16) ? 16 : 0) + hh;
    ws[i] = (_Float16)W[(N * kdim + Kk) * 3 + 1];   // center tap only
}

// Fast gate nonlinearities: single v_exp_f32 + v_rcp_f32 each.
__device__ __forceinline__ float sigf(float x) {
    float e = __builtin_amdgcn_exp2f(-1.4426950408889634f * x);
    return __builtin_amdgcn_rcpf(1.0f + e);
}
__device__ __forceinline__ float tanh_fast(float x) {
    float e = __builtin_amdgcn_exp2f(-2.8853900817779268f * x);
    return 2.0f * __builtin_amdgcn_rcpf(1.0f + e) - 1.0f;
}

// ---------------------------------------------------------------------------
// Main persistent kernel: 16 blocks (one 16-row batch tile each) x 512 threads
// (16 wave32s). Whole T=512 recurrence runs on-chip; weights stream from L2 as
// prebuilt WMMA B-fragments; activations/gates go through LDS; c-state lives
// in registers with a fixed thread<->(m,j) mapping.
// ---------------------------------------------------------------------------
__global__ void __launch_bounds__(512)
convlstm_main_kernel(const float* __restrict__ x,
                     const float* __restrict__ b1, const float* __restrict__ b2,
                     const float* __restrict__ b3,
                     const float* __restrict__ h1i, const float* __restrict__ c1i,
                     const float* __restrict__ h2i, const float* __restrict__ c2i,
                     const float* __restrict__ h3i, const float* __restrict__ c3i,
                     const _Float16* __restrict__ ws,
                     const int* __restrict__ seq_len_p,
                     float* __restrict__ out)
{
    __shared__ __align__(16) _Float16 act[16][ACT_STRIDE]; // cols: x[0,64) h1[64,192) h2[192,320) h3[320,448)
    __shared__ float cc[16][CC_STRIDE];                    // gate pre-activations, 16 x 512
    __shared__ float bias_s[3][NG];

    const int tid   = threadIdx.x;
    const int tile  = blockIdx.x;           // 16 batch rows per tile
    const int wave  = tid >> 5;
    const int lane  = tid & 31;
    const int mrow  = lane & 15;            // A-matrix row for this lane
    const int khalf = (lane >> 4) * 16;     // K sub-offset: contiguous 16 K per lane half (matches B frags)
    const int mbase = (lane >= 16) ? 8 : 0; // C/D row base per ISA layout
    const int ncol  = lane & 15;            // C/D column within tile

    // Biases into LDS (one-time)
    for (int i = tid; i < NG; i += 512) {
        bias_s[0][i] = b1[i]; bias_s[1][i] = b2[i]; bias_s[2][i] = b3[i];
    }
    // Initial hidden states into LDS activation buffer (f16)
    for (int i = tid; i < 16 * H; i += 512) {
        int m = i >> 7, j = i & 127;
        int g = (tile * 16 + m) * H + j;
        act[m][64          + j] = (_Float16)h1i[g];
        act[m][64 + H      + j] = (_Float16)h2i[g];
        act[m][64 + 2 * H  + j] = (_Float16)h3i[g];
    }
    // Cell state in registers: thread handles gate elements e = rep*512+tid forever
    float creg[3][4];
#pragma unroll
    for (int rep = 0; rep < 4; ++rep) {
        int e = rep * 512 + tid; int m = e >> 7, j = e & 127;
        int g = (tile * 16 + m) * H + j;
        creg[0][rep] = c1i[g]; creg[1][rep] = c2i[g]; creg[2][rep] = c3i[g];
    }
    const int Trun = *seq_len_p;
    __syncthreads();

    for (int t = 0; t < Trun; ++t) {
        // Opaque copy of the weight base each iteration: forbids LICM from
        // hoisting the (t-invariant) B-fragment loads out of the time loop
        // (which previously caused VGPR spills). Rebuilt as an explicit
        // address_space(1) pointer so the backend emits global_load_b128
        // (LOADcnt only), not flat_load (which also ties up DScnt).
        uintptr_t wsb = (uintptr_t)ws;
        asm volatile("" : "+s"(wsb));
        gwhalf wst = (gwhalf)wsb;

        // Load x tile (16 x 64) -> act f16
#pragma unroll
        for (int rep = 0; rep < 2; ++rep) {
            int e = rep * 512 + tid; int m = e >> 6, c = e & 63;
            act[m][c] = (_Float16)x[((tile * 16 + m) * T_ + t) * CIN + c];
        }
        __syncthreads();

#pragma unroll
        for (int L = 0; L < 3; ++L) {
            const int nk    = (L == 0) ? NK1 : NK23;
            const int a_off = (L == 0) ? 0 : ((L == 1) ? 64 : 64 + H);
            const int h_off = 64 + L * H;
            const int wbase = (L == 0) ? WS_L1 : ((L == 1) ? WS_L2 : WS_L3);

            // ---- A fragments (16x32 f16) from LDS, contiguous 16 K per lane half via 2x ds_load_b128
            v16h afrag[8];
#pragma unroll
            for (int kt = 0; kt < 8; ++kt) {
                if (kt < nk) {
                    const int cb = a_off + kt * 32 + khalf;
                    union { v16h v; v8h h[2]; } u;
                    u.h[0] = *(const v8h*)&act[mrow][cb];
                    u.h[1] = *(const v8h*)&act[mrow][cb + 8];
                    afrag[kt] = u.v;
                }
            }
            // ---- each wave computes 2 N-tiles of 16 gate columns
#pragma unroll
            for (int nt = 0; nt < 2; ++nt) {
                const int ntile = wave * 2 + nt;
                gwhalf wf = wst + wbase + (ntile * nk) * FRAG_H + lane * 16;
                v8f acc = {};
#pragma unroll
                for (int kt = 0; kt < 8; ++kt) {
                    if (kt < nk) {
                        v16h bfrag = *(gv16hp)(wf + kt * FRAG_H);
                        acc = __builtin_amdgcn_wmma_f32_16x16x32_f16(
                                  false, afrag[kt], false, bfrag,
                                  (short)0, acc, false, false);
                    }
                }
                const int n0 = ntile * 16 + ncol;
#pragma unroll
                for (int r = 0; r < 8; ++r) cc[mbase + r][n0] = acc[r];
            }
            __syncthreads();

            // ---- fused gate math; c-state in registers, h back to LDS f16
            const bool last = (L == 2) && (t == Trun - 1);
#pragma unroll
            for (int rep = 0; rep < 4; ++rep) {
                int e = rep * 512 + tid; int m = e >> 7, j = e & 127;
                float vi = cc[m][j          ] + bias_s[L][j          ];
                float vf = cc[m][j + H      ] + bias_s[L][j + H      ];
                float vo = cc[m][j + 2 * H  ] + bias_s[L][j + 2 * H  ];
                float vg = cc[m][j + 3 * H  ] + bias_s[L][j + 3 * H  ];
                float cn = sigf(vf) * creg[L][rep] + sigf(vi) * tanh_fast(vg);
                float hn = sigf(vo) * tanh_fast(cn);
                creg[L][rep] = cn;
                act[m][h_off + j] = (_Float16)hn;
                if (last) out[(tile * 16 + m) * H + j] = hn;
            }
            __syncthreads();
        }
    }
}

extern "C" void kernel_launch(void* const* d_in, const int* in_sizes, int n_in,
                              void* d_out, int out_size, void* d_ws, size_t ws_size,
                              hipStream_t stream)
{
    (void)in_sizes; (void)n_in; (void)out_size; (void)ws_size;
    const float* x  = (const float*)d_in[0];
    const float* W1 = (const float*)d_in[1];
    const float* b1 = (const float*)d_in[2];
    const float* W2 = (const float*)d_in[3];
    const float* b2 = (const float*)d_in[4];
    const float* W3 = (const float*)d_in[5];
    const float* b3 = (const float*)d_in[6];
    const float* h1 = (const float*)d_in[7];
    const float* c1 = (const float*)d_in[8];
    const float* h2 = (const float*)d_in[9];
    const float* c2 = (const float*)d_in[10];
    const float* h3 = (const float*)d_in[11];
    const float* c3 = (const float*)d_in[12];
    const int* seq_len = (const int*)d_in[13];
    float* out = (float*)d_out;
    _Float16* ws = (_Float16*)d_ws;   // 720,896 bytes of f16 WMMA B-fragments

    // Pass 1: build WMMA weight fragments (L2-resident afterwards)
    prep_weights_kernel<<<(WS_TOT + 511) / 512, 512, 0, stream>>>(W1, W2, W3, ws);

    // Pass 2: persistent recurrence, one workgroup per 16-row batch tile
    convlstm_main_kernel<<<B_ / 16, 512, 0, stream>>>(
        x, b1, b2, b3, h1, c1, h2, c2, h3, c3, ws, seq_len, out);
}
